// DynamicWeightAttention_12214886990207
// MI455X (gfx1250) — matrix operand order repaired and checked by reading, not compiled, not run on hardware
//
#include <hip/hip_runtime.h>

typedef __attribute__((ext_vector_type(16))) _Float16 v16h;
typedef __attribute__((ext_vector_type(8)))  float    v8f;

#define HH  16   // HRU dim (= softmax group = WMMA M)
#define PD  16
#define PS  16
#define CF  32   // fused feature dim (= WMMA K)
#define HID 64   // hidden (= 4 WMMA N tiles)
#define TILES_PER_WAVE 16

__device__ __forceinline__ float fast_tanh(float x) {
#if __has_builtin(__builtin_amdgcn_tanhf)
    return __builtin_amdgcn_tanhf(x);          // native v_tanh_f32 (gfx1250 trans op)
#else
    // tanh(x) = 1 - 2/(exp(2x)+1); exp(2x) = exp2(x * 2*log2(e))
    float t = __builtin_amdgcn_exp2f(x * 2.885390081777927f);
    float r = __builtin_amdgcn_rcpf(t + 1.0f);
    return 1.0f - 2.0f * r;
#endif
}

__global__ __launch_bounds__(256) void dwa_kernel(
    const float* __restrict__ dyn,   const float* __restrict__ stat,
    const float* __restrict__ gamma, const float* __restrict__ beta,
    const float* __restrict__ w1,    const float* __restrict__ b1,
    const float* __restrict__ w2,    float* __restrict__ out,
    int numTiles, int N)
{
    // per-wave scratch: 16 rows x 20 floats (pad keeps banks conflict-free,
    // rows 16B-aligned for ds_load_b128)
    __shared__ float scr[8][16 * 20];

    const int tid  = threadIdx.x;
    const int wave = tid >> 5;
    const int lane = tid & 31;
    const int grp  = lane >> 4;   // 0: K low half / M 0-7 ; 1: K high half / M 8-15
    const int m    = lane & 15;

    // ---- B fragments: w1 (32x64) as 4 tiles of 32x16, f16 ----
    // B layout: lane holds N = m (+16t), K = j<8 ? grp*8 + j : 16 + grp*8 + (j-8)  (j = packed half idx)
    v16h bfrag[4];
#pragma unroll
    for (int t = 0; t < 4; ++t) {
#pragma unroll
        for (int j = 0; j < 16; ++j)
            bfrag[t][j] = (_Float16)w1[((j & 8) * 2 + grp * 8 + (j & 7)) * HID + t * 16 + m];
    }
    float w2v[4], b1v[4];
#pragma unroll
    for (int t = 0; t < 4; ++t) { w2v[t] = w2[t * 16 + m]; b1v[t] = b1[t * 16 + m]; }

    // ---- per-lane gamma/beta for this lane's 16 fused features ----
    // j<8 : k = grp*8+j (dynamic part); j>=8 : k = 16 + grp*8 + (j-8) (static part)
    float ga[16], bb[16];
#pragma unroll
    for (int j = 0; j < 8; ++j) {
        ga[j]     = gamma[grp * 8 + j];       bb[j]     = beta[grp * 8 + j];
        ga[j + 8] = gamma[16 + grp * 8 + j];  bb[j + 8] = beta[16 + grp * 8 + j];
    }

    int tile = blockIdx.x * 8 + wave;
    const int stride = gridDim.x * 8;
    const int laneOff = m * PD + grp * 8;

    for (; tile < numTiles; tile += stride) {
        const int n = tile % N;
        const float* dp = dyn  + (size_t)tile * (HH * PD) + laneOff;
        const float* sp = stat + (size_t)n    * (HH * PS) + m * PS + grp * 8;

        // prefetch next tile's dynamic block (grid-stride address is known)
        if (tile + stride < numTiles)
            __builtin_prefetch(dyn + (size_t)(tile + stride) * (HH * PD) + laneOff, 0, 1);

        // lane's 16 fused features: [dyn grp*8..+7 | stat grp*8..+7]
        float4 d0 = *(const float4*)(dp);
        float4 d1 = *(const float4*)(dp + 4);
        float4 s0 = *(const float4*)(sp);
        float4 s1 = *(const float4*)(sp + 4);

        float v[16];
        v[0]=d0.x; v[1]=d0.y; v[2]=d0.z; v[3]=d0.w;
        v[4]=d1.x; v[5]=d1.y; v[6]=d1.z; v[7]=d1.w;
        v[8]=s0.x; v[9]=s0.y; v[10]=s0.z; v[11]=s0.w;
        v[12]=s1.x; v[13]=s1.y; v[14]=s1.z; v[15]=s1.w;

        // ---- LayerNorm over 32 feats (biased var, eps=1e-4) ----
        float sum = 0.f, sq = 0.f;
#pragma unroll
        for (int j = 0; j < 16; ++j) { sum += v[j]; sq += v[j] * v[j]; }
        sum += __shfl_xor(sum, 16, 32);   // combine with partner half of row
        sq  += __shfl_xor(sq, 16, 32);
        float mean = sum * (1.0f / 32.0f);
        float var  = sq  * (1.0f / 32.0f) - mean * mean;
        float rstd = __builtin_amdgcn_rsqf(var + 1e-4f);

        // ---- normalize + pack A fragment (16-bit A 16x32 layout) ----
        v16h a;
#pragma unroll
        for (int j = 0; j < 16; ++j)
            a[j] = (_Float16)((v[j] - mean) * rstd * ga[j] + bb[j]);

        // ---- 4x WMMA (C seeded with b1), tanh, dot with w2 ----
        float s[8];
#pragma unroll
        for (int r = 0; r < 8; ++r) s[r] = 0.f;
#pragma unroll
        for (int t = 0; t < 4; ++t) {
            v8f c;
#pragma unroll
            for (int r = 0; r < 8; ++r) c[r] = b1v[t];
            c = __builtin_amdgcn_wmma_f32_16x16x32_f16(
                    false, a, false, bfrag[t], (short)0, c, false, false);
#pragma unroll
            for (int r = 0; r < 8; ++r) s[r] += fast_tanh(c[r]) * w2v[t];
        }

        // ---- transpose partial scores through LDS: row M = r + grp*8 ----
        float* sc = scr[wave];
#pragma unroll
        for (int r = 0; r < 8; ++r) sc[(r + grp * 8) * 20 + m] = s[r];
        asm volatile("s_wait_dscnt 0x0" ::: "memory");

        float4 p0 = *(const float4*)(sc + m * 20);
        float4 p1 = *(const float4*)(sc + m * 20 + 4);
        float4 p2 = *(const float4*)(sc + m * 20 + 8);
        float4 p3 = *(const float4*)(sc + m * 20 + 12);
        float score = ((p0.x + p0.y) + (p0.z + p0.w)) + ((p1.x + p1.y) + (p1.z + p1.w))
                    + ((p2.x + p2.y) + (p2.z + p2.w)) + ((p3.x + p3.y) + (p3.z + p3.w));
        // (b2 is constant across the softmax group -> cancels; omitted)

        // ---- softmax over the 16 rows (butterfly within 16-lane groups) ----
        float mx = score;
#pragma unroll
        for (int msk = 1; msk < 16; msk <<= 1)
            mx = fmaxf(mx, __shfl_xor(mx, msk, 32));
        float e = __builtin_amdgcn_exp2f((score - mx) * 1.4426950408889634f);
        float tot = e;
#pragma unroll
        for (int msk = 1; msk < 16; msk <<= 1)
            tot += __shfl_xor(tot, msk, 32);
        float res = e * __builtin_amdgcn_rcpf(tot);

        if (lane < 16)
            out[(size_t)tile * HH + m] = res;
    }
}

extern "C" void kernel_launch(void* const* d_in, const int* in_sizes, int n_in,
                              void* d_out, int out_size, void* d_ws, size_t ws_size,
                              hipStream_t stream) {
    const float* dyn   = (const float*)d_in[0];
    const float* stat  = (const float*)d_in[1];
    const float* gamma = (const float*)d_in[2];
    const float* beta  = (const float*)d_in[3];
    const float* w1    = (const float*)d_in[4];
    const float* b1    = (const float*)d_in[5];
    const float* w2    = (const float*)d_in[6];
    // d_in[7] = b2: constant shift before softmax -> cancels exactly; unused.

    const int numTiles = in_sizes[0] / (HH * PD);   // B_T * N
    const int N        = in_sizes[1] / (HH * PS);

    // 8 waves/block, one tile per wave per iteration, ~16 iterations per wave
    int blocks = (numTiles + 8 * TILES_PER_WAVE - 1) / (8 * TILES_PER_WAVE);
    if (blocks < 1) blocks = 1;

    dwa_kernel<<<blocks, 256, 0, stream>>>(dyn, stat, gamma, beta, w1, b1, w2,
                                           (float*)d_out, numTiles, N);
}
